// GAT_5755256177064
// MI455X (gfx1250) — matrix operand order repaired and checked by reading, not compiled, not run on hardware
//
#include <hip/hip_runtime.h>
#include <math.h>

// ---------------------------------------------------------------------------
// GAT 2-layer forward for MI455X (gfx1250, wave32).
//  * GEMMs: v_wmma_f32_16x16x32_bf16, operands pre-converted to padded bf16,
//    tiles streamed LDS-double-buffered via global_load_async_to_lds_b128
//    (+ s_wait_asynccnt), zero guards in the K loop.
//  * Edge softmax/aggregation via L2 atomics (memory-bound phase).
// ---------------------------------------------------------------------------

#define IN_F   1433
#define KP1    1440          // IN_F padded to multiple of 32
#define KP2    256
#define NP1    256
#define NP2    64            // 56 cols padded to 64
#define HEADS  8
#define HID    32
#define NCLS   7
#define SLOPE  0.2f

typedef __attribute__((ext_vector_type(8)))  float          v8f;
typedef __attribute__((ext_vector_type(16))) __bf16         v16bf;
typedef __attribute__((ext_vector_type(8)))  unsigned short us8;
typedef __attribute__((ext_vector_type(16))) unsigned short us16;

static __device__ __forceinline__ unsigned short f2bf(float f) {
    unsigned u = __float_as_uint(f);
    u += 0x7fffu + ((u >> 16) & 1u);          // round-to-nearest-even
    return (unsigned short)(u >> 16);
}

static __device__ __forceinline__ v16bf pack_frag(us8 lo, us8 hi) {
    us16 w = __builtin_shufflevector(lo, hi, 0,1,2,3,4,5,6,7,8,9,10,11,12,13,14,15);
    return __builtin_bit_cast(v16bf, w);
}

// generic pointer -> 32-bit LDS byte address (addrspace(3) offset)
static __device__ __forceinline__ unsigned lds_addr(const void* p) {
    return (unsigned)(unsigned long long)
           (__attribute__((address_space(3))) const void*)p;
}

// one 16-byte async chunk: global -> LDS (tracked by ASYNCcnt)
static __device__ __forceinline__ void async_b128(unsigned lds, const void* gptr) {
    asm volatile("global_load_async_to_lds_b128 %0, %1, off"
                 :: "v"(lds), "v"(gptr) : "memory");
}
static __device__ __forceinline__ void wait_async() {
    asm volatile("s_wait_asynccnt 0" ::: "memory");
}

// ---------------------------------------------------------------------------
// pre-conversion kernels (run once, ~0.4 GB total traffic)
// ---------------------------------------------------------------------------
__global__ void cvt_pad_bf16(const float* __restrict__ src,
                             unsigned short* __restrict__ dst,
                             int rows, int K, int KP) {
    long long i = (long long)blockIdx.x * blockDim.x + threadIdx.x;
    if (i >= (long long)rows * KP) return;
    int r = (int)(i / KP), k = (int)(i - (long long)r * KP);
    dst[i] = (k < K) ? f2bf(src[(size_t)r * K + k]) : 0;
}

// W[K x Ncols] f32 -> WT[NP x KP] bf16 (transposed, zero padded)
__global__ void transpose_bf16(const float* __restrict__ W,
                               unsigned short* __restrict__ WT,
                               int K, int Ncols, int KP, int NP) {
    long long i = (long long)blockIdx.x * blockDim.x + threadIdx.x;
    if (i >= (long long)NP * KP) return;
    int n = (int)(i / KP), k = (int)(i - (long long)n * KP);
    WT[i] = (n < Ncols && k < K) ? f2bf(W[(size_t)k * Ncols + n]) : 0;
}

// ---------------------------------------------------------------------------
// Async double-buffered bf16 WMMA GEMM:
//   C[M x ldC] (f32) = A[M x KP] (bf16) * BT[NP x KP]^T (bf16)
// 128x64 tile / 256-thread block; 8 waves x (16 rows, 4 col sub-tiles).
// KP multiple of 32, gridDim.y*64 == NP: no guards in the hot loop.
// ---------------------------------------------------------------------------
__global__ __launch_bounds__(256)
void gemm_async_wmma(const unsigned short* __restrict__ A,
                     const unsigned short* __restrict__ BT,
                     float* __restrict__ C,
                     int M, int KP, int Ncols, int ldC) {
    // row stride 40 halfs = 80 B: every 8-half chunk is 16B aligned, and the
    // 16 fragment row-lanes hit distinct banks (m*20 mod 64 all distinct).
    __shared__ __align__(16) unsigned short As[2][128][40];
    __shared__ __align__(16) unsigned short Bt[2][64][40];

    const int t    = threadIdx.x;
    const int lane = t & 31;
    const int wave = t >> 5;
    const int m0   = blockIdx.x * 128;
    const int n0   = blockIdx.y * 64;

    v8f acc[4];
#pragma unroll
    for (int s = 0; s < 4; ++s) acc[s] = (v8f){0,0,0,0,0,0,0,0};

    const int row_half = lane & 15;   // M (A frag) / N (B frag) index
    const int half     = lane >> 4;   // K-half selector

    // ---- async tile staging (3 x b128 issues per thread per K-step) ----
    auto stageA = [&](int buf, int k0) {
#pragma unroll
        for (int i = 0; i < 2; ++i) {
            int u   = t + 256 * i;           // 512 chunks: 128 rows x 4 groups
            int row = u >> 2, grp = u & 3;
            int gm  = m0 + row; gm = gm < M ? gm : M - 1;   // clamp, no zero fill
            async_b128(lds_addr(&As[buf][row][grp * 8]),
                       A + (size_t)gm * KP + k0 + grp * 8);
        }
    };
    auto stageB = [&](int buf, int k0) {
        int n = t >> 2, grp = t & 3;         // 256 chunks: 64 rows x 4 groups
        async_b128(lds_addr(&Bt[buf][n][grp * 8]),
                   BT + (size_t)(n0 + n) * KP + k0 + grp * 8);
    };

    stageA(0, 0);
    stageB(0, 0);

    const int NT = KP >> 5;
    for (int kt = 0; kt < NT; ++kt) {
        const int cur = kt & 1;
        wait_async();                 // this wave's current tile landed in LDS
        __syncthreads();              // everyone's tile visible
        if (kt + 1 < NT) {            // stream next tile while we compute
            stageA(cur ^ 1, (kt + 1) * 32);
            stageB(cur ^ 1, (kt + 1) * 32);
        }

        // A fragment (16x32 bf16, ISA 7.12.2): VGPR0-3 K=half*8+j, VGPR4-7 +16
        const unsigned short* arow = &As[cur][wave * 16 + row_half][0];
        us8 alo = *(const us8*)(arow + half * 8);
        us8 ahi = *(const us8*)(arow + 16 + half * 8);
        v16bf afrag = pack_frag(alo, ahi);

#pragma unroll
        for (int s = 0; s < 4; ++s) {
            // B fragment (32x16 bf16): same K order as A: K=half*8+j, then +16
            const unsigned short* brow = &Bt[cur][s * 16 + row_half][0];
            us8 blo = *(const us8*)(brow + half * 8);
            us8 bhi = *(const us8*)(brow + 16 + half * 8);
            acc[s] = __builtin_amdgcn_wmma_f32_16x16x32_bf16(
                false, afrag, false, pack_frag(blo, bhi), (short)0,
                acc[s], false, false);
        }
        __syncthreads();              // all reads of `cur` done before reuse
    }

    // ---- epilogue: lane(0-15)->M=r, lane(16-31)->M=8+r, N=lane&15 ----
#pragma unroll
    for (int s = 0; s < 4; ++s) {
#pragma unroll
        for (int r = 0; r < 8; ++r) {
            int gm = m0 + wave * 16 + half * 8 + r;
            int gn = n0 + s * 16 + row_half;
            if (gm < M && gn < Ncols) C[(size_t)gm * ldC + gn] = acc[s][r];
        }
    }
}

// ---------------------------------------------------------------------------
// attention phase (memory/atomic bound)
// ---------------------------------------------------------------------------
__global__ void alpha_kernel(const float* __restrict__ H,
                             const float* __restrict__ att_s,
                             const float* __restrict__ att_d,
                             float* __restrict__ asrc, float* __restrict__ adst,
                             int Nn, int C, int CT) {
    int idx = blockIdx.x * blockDim.x + threadIdx.x;
    if (idx >= Nn * HEADS) return;
    int n = idx >> 3, h = idx & 7;
    const float* row = H + (size_t)n * CT + h * C;
    float ss = 0.f, sd = 0.f;
    for (int c = 0; c < C; ++c) {
        float v = row[c];
        ss += v * att_s[h * C + c];
        sd += v * att_d[h * C + c];
    }
    asrc[idx] = ss;
    adst[idx] = sd;
}

__global__ void init_kernel(float* __restrict__ m, float* __restrict__ den,
                            float* __restrict__ acc, int Nn, int CT) {
    int i = blockIdx.x * blockDim.x + threadIdx.x;
    if (i < Nn * HEADS) { m[i] = -__builtin_inff(); den[i] = 0.f; }
    if (i < Nn * CT)    acc[i] = 0.f;
}

static __device__ __forceinline__ void atomicMaxF(float* addr, float v) {
    if (v >= 0.f) atomicMax((int*)addr, __float_as_int(v));
    else          atomicMin((unsigned int*)addr, __float_as_uint(v));
}

static __device__ __forceinline__ void edge_sd(const int* __restrict__ ei,
                                               int e, int E, int& s, int& d) {
    if (e < E) { s = ei[e]; d = ei[E + e]; }
    else       { s = d = e - E; }            // self loop
}

__global__ void emax_kernel(const int* __restrict__ ei,
                            const float* __restrict__ asrc,
                            const float* __restrict__ adst,
                            float* __restrict__ m, int E, int Nn) {
    int idx = blockIdx.x * blockDim.x + threadIdx.x;
    if (idx >= (E + Nn) * HEADS) return;
    int h = idx & 7, e = idx >> 3, s, d;
    edge_sd(ei, e, E, s, d);
    float v = asrc[s * HEADS + h] + adst[d * HEADS + h];
    v = v > 0.f ? v : SLOPE * v;
    atomicMaxF(&m[d * HEADS + h], v);
}

__global__ void eden_kernel(const int* __restrict__ ei,
                            const float* __restrict__ asrc,
                            const float* __restrict__ adst,
                            const float* __restrict__ m,
                            float* __restrict__ den, int E, int Nn) {
    int idx = blockIdx.x * blockDim.x + threadIdx.x;
    if (idx >= (E + Nn) * HEADS) return;
    int h = idx & 7, e = idx >> 3, s, d;
    edge_sd(ei, e, E, s, d);
    float v = asrc[s * HEADS + h] + adst[d * HEADS + h];
    v = v > 0.f ? v : SLOPE * v;
    atomicAdd(&den[d * HEADS + h], __expf(v - m[d * HEADS + h]));
}

__global__ void eagg_kernel(const int* __restrict__ ei,
                            const float* __restrict__ asrc,
                            const float* __restrict__ adst,
                            const float* __restrict__ m,
                            const float* __restrict__ H,
                            float* __restrict__ acc,
                            int E, int Nn, int C, int CT) {
    int e = blockIdx.x;
    int c = threadIdx.x;
    if (c >= CT) return;
    int s, d;
    edge_sd(ei, e, E, s, d);
    int h = c / C;
    float v = asrc[s * HEADS + h] + adst[d * HEADS + h];
    v = v > 0.f ? v : SLOPE * v;
    float p = __expf(v - m[d * HEADS + h]);
    atomicAdd(&acc[(size_t)d * CT + c], p * H[(size_t)s * CT + c]);
}

// normalize + bias (+ELU); optionally emits bf16 copy (next layer's A operand)
__global__ void fin_kernel(const float* __restrict__ acc,
                           const float* __restrict__ den,
                           const float* __restrict__ bias,
                           float* __restrict__ outf,
                           unsigned short* __restrict__ outb,
                           int Nn, int C, int CT, int do_elu) {
    int i = blockIdx.x * blockDim.x + threadIdx.x;
    if (i >= Nn * CT) return;
    int n = i / CT, c = i - n * CT, h = c / C;
    float v = acc[i] / den[n * HEADS + h] + bias[c];
    if (do_elu) v = v > 0.f ? v : (__expf(v) - 1.0f);
    if (outf) outf[i] = v;
    if (outb) outb[i] = f2bf(v);
}

// ---------------------------------------------------------------------------
extern "C" void kernel_launch(void* const* d_in, const int* in_sizes, int n_in,
                              void* d_out, int out_size, void* d_ws, size_t ws_size,
                              hipStream_t stream) {
    const float* x   = (const float*)d_in[0];
    const int*   ei  = (const int*)  d_in[1];
    const float* W1  = (const float*)d_in[2];
    const float* as1 = (const float*)d_in[3];
    const float* ad1 = (const float*)d_in[4];
    const float* b1  = (const float*)d_in[5];
    const float* W2  = (const float*)d_in[6];
    const float* as2 = (const float*)d_in[7];
    const float* ad2 = (const float*)d_in[8];
    const float* b2  = (const float*)d_in[9];
    float* out = (float*)d_out;

    const int Nn  = in_sizes[0] / IN_F;         // 50000
    const int E   = in_sizes[1] / 2;            // 800000
    const int ET  = E + Nn;                     // + self loops
    const int C1T = HEADS * HID;                // 256
    const int C2T = HEADS * NCLS;               // 56

    // ---- workspace layout (all slot offsets 16B aligned) ----
    float* H    = (float*)d_ws;                 // [N,256] f32 (L2 reuses ld=56)
    float* asr  = H   + (size_t)Nn * C1T;       // [N,8]
    float* ads  = asr + (size_t)Nn * HEADS;
    float* mm   = ads + (size_t)Nn * HEADS;
    float* den  = mm  + (size_t)Nn * HEADS;
    float* acc  = den + (size_t)Nn * HEADS;     // [N,256]
    unsigned short* Xbf  = (unsigned short*)(acc + (size_t)Nn * C1T); // [N,KP1]
    unsigned short* Hbf  = Xbf + (size_t)Nn * KP1;                    // [N,KP2]
    unsigned short* W1T  = Hbf + (size_t)Nn * KP2;                    // [NP1,KP1]
    unsigned short* W2T  = W1T + (size_t)NP1 * KP1;                   // [NP2,KP2]

    const int mtiles = (Nn + 127) / 128;
    dim3 blk(256);

    // ---- one-time operand conversion to padded bf16 ----
    cvt_pad_bf16<<<(int)(((long long)Nn * KP1 + 255) / 256), blk, 0, stream>>>(
        x, Xbf, Nn, IN_F, KP1);
    transpose_bf16<<<(NP1 * KP1 + 255) / 256, blk, 0, stream>>>(
        W1, W1T, IN_F, C1T, KP1, NP1);
    transpose_bf16<<<(NP2 * KP2 + 255) / 256, blk, 0, stream>>>(
        W2, W2T, C1T, C2T, KP2, NP2);

    // ===================== layer 1 =====================
    gemm_async_wmma<<<dim3(mtiles, NP1 / 64), blk, 0, stream>>>(
        Xbf, W1T, H, Nn, KP1, C1T, C1T);
    alpha_kernel<<<(Nn * HEADS + 255) / 256, blk, 0, stream>>>(
        H, as1, ad1, asr, ads, Nn, HID, C1T);
    init_kernel<<<(Nn * C1T + 255) / 256, blk, 0, stream>>>(mm, den, acc, Nn, C1T);
    emax_kernel<<<(ET * HEADS + 255) / 256, blk, 0, stream>>>(ei, asr, ads, mm, E, Nn);
    eden_kernel<<<(ET * HEADS + 255) / 256, blk, 0, stream>>>(ei, asr, ads, mm, den, E, Nn);
    eagg_kernel<<<ET, 256, 0, stream>>>(ei, asr, ads, mm, H, acc, E, Nn, HID, C1T);
    fin_kernel<<<(Nn * C1T + 255) / 256, blk, 0, stream>>>(
        acc, den, b1, nullptr, Hbf, Nn, HID, C1T, /*elu=*/1);

    // ===================== layer 2 =====================
    gemm_async_wmma<<<dim3(mtiles, NP2 / 64), blk, 0, stream>>>(
        Hbf, W2T, H, Nn, KP2, C2T, C2T);
    alpha_kernel<<<(Nn * HEADS + 255) / 256, blk, 0, stream>>>(
        H, as2, ad2, asr, ads, Nn, NCLS, C2T);
    init_kernel<<<(Nn * C2T + 255) / 256, blk, 0, stream>>>(mm, den, acc, Nn, C2T);
    emax_kernel<<<(ET * HEADS + 255) / 256, blk, 0, stream>>>(ei, asr, ads, mm, E, Nn);
    eden_kernel<<<(ET * HEADS + 255) / 256, blk, 0, stream>>>(ei, asr, ads, mm, den, E, Nn);
    eagg_kernel<<<ET, 64, 0, stream>>>(ei, asr, ads, mm, H, acc, E, Nn, NCLS, C2T);
    fin_kernel<<<(Nn * C2T + 255) / 256, blk, 0, stream>>>(
        acc, den, b2, out, nullptr, Nn, NCLS, C2T, /*elu=*/0);
}
